// MPMultiHeadAttention_88613765251733
// MI455X (gfx1250) — matrix operand order repaired and checked by reading, not compiled, not run on hardware
//
#include <hip/hip_runtime.h>
#include <math.h>

// Problem constants (match reference)
#define B_  2
#define L_  1024
#define D_  512
#define H_  8
#define DK_ 64
#define DV_ 64
#define P_  4

// Flip to 0 if the assembler rejects the async-to-LDS mnemonics.
#ifndef USE_ASYNC_LDS
#define USE_ASYNC_LDS 1
#endif

typedef __attribute__((ext_vector_type(16))) _Float16 v16h;
typedef __attribute__((ext_vector_type(8)))  float    v8f;

// ---------------- WMMA fragment helpers (wave32, 16x16x32 f16) ----------------

// A fragment: 16x32 (MxK), row-major source, ld = row stride.
// Lane l holds row (l&15); K packing: kk = (l>>4)*16 + 2v (16 contiguous halfs).
__device__ __forceinline__ v16h frag_a(const _Float16* __restrict__ p, int ld, int lane) {
  v16h f;
  const int row = lane & 15, half = lane >> 4;
  const _Float16* q = p + (size_t)row * ld + half * 16;
#pragma unroll
  for (int v = 0; v < 8; ++v) {
    const int kk = v << 1;
    f[2 * v]     = q[kk];
    f[2 * v + 1] = q[kk + 1];
  }
  return f;
}

// B fragment (NT): 32x16 (KxN) where source is row-major [N,K].
// Lane l holds column (l&15); K packing: kk = (l>>4)*16 + 2v (16 contiguous halfs).
__device__ __forceinline__ v16h frag_b_nt(const _Float16* __restrict__ p, int ld, int lane) {
  v16h f;
  const int col = lane & 15, half = lane >> 4;
  const _Float16* q = p + (size_t)col * ld + half * 16;
#pragma unroll
  for (int v = 0; v < 8; ++v) {
    f[2 * v]     = q[2 * v];
    f[2 * v + 1] = q[2 * v + 1];
  }
  return f;
}

__device__ __forceinline__ v8f wmma16(v16h a, v16h b, v8f c) {
  return __builtin_amdgcn_wmma_f32_16x16x32_f16(false, a, false, b, (short)0, c, false, false);
}

// ---------------- Async 16B copy global -> LDS (CDNA5 ASYNCcnt path) ----------------
__device__ __forceinline__ void stage16(const _Float16* g, _Float16* l) {
#if USE_ASYNC_LDS
  unsigned laddr = (unsigned)(uintptr_t)l;                 // low 32 bits = LDS offset
  unsigned long long ga = (unsigned long long)(uintptr_t)g;
  asm volatile("global_load_async_to_lds_b128 %0, %1, off"
               :: "v"(laddr), "v"(ga) : "memory");
#else
  *(float4*)l = *(const float4*)g;
#endif
}

__device__ __forceinline__ void stage_wait() {
#if USE_ASYNC_LDS
  asm volatile("s_wait_asynccnt 0x0" ::: "memory");
#endif
}

// ---------------- Shared GEMM body: A from global (per-wave row tile),
// B supertile (64 rows x K) double-buffered in LDS, 4 WMMAs per K-step ----------
// lds must hold 2 * 64 * 32 halfs (8 KB).
__device__ __forceinline__ void gemm_nt_lds(const _Float16* __restrict__ Arow, int lda,
                                            const _Float16* __restrict__ Bsup, int ldb,
                                            int K, _Float16* lds, v8f acc[4]) {
  const int tid  = threadIdx.x;           // 256 threads
  const int lane = tid & 31;
  const int trow = tid >> 2;              // 0..63: B row within supertile
  const int tcol = (tid & 3) * 8;         // 0,8,16,24: K offset (16B chunks)
  const int nsteps = K >> 5;

  // Prologue: stage K-step 0 into buffer 0.
  stage16(Bsup + (size_t)trow * ldb + tcol, lds + trow * 32 + tcol);

  for (int ks = 0; ks < nsteps; ++ks) {
    _Float16* cur = lds + (ks & 1) * (64 * 32);
    stage_wait();
    __syncthreads();                      // staged tile visible to all waves
    if (ks + 1 < nsteps) {
      _Float16* nxt = lds + ((ks + 1) & 1) * (64 * 32);
      stage16(Bsup + (size_t)trow * ldb + (ks + 1) * 32 + tcol, nxt + trow * 32 + tcol);
    }
    __builtin_prefetch(Arow + (ks + 2) * 32, 0, 1);
    v16h a = frag_a(Arow + ks * 32, lda, lane);
#pragma unroll
    for (int j = 0; j < 4; ++j) {
      v16h b = frag_b_nt(cur + (16 * j) * 32, 32, lane);
      acc[j] = wmma16(a, b, acc[j]);
    }
  }
}

// ---------------- LayerNorm(q) -> f16 ----------------
__global__ void __launch_bounds__(128)
ln_kernel(const float* __restrict__ x, const float* __restrict__ g,
          const float* __restrict__ b, _Float16* __restrict__ out) {
  const int row = blockIdx.x;            // B*L rows
  const int tid = threadIdx.x;           // 128
  const float* xr = x + (size_t)row * D_;
  float s = 0.f, ss = 0.f;
#pragma unroll
  for (int i = tid; i < D_; i += 128) { float v = xr[i]; s += v; ss += v * v; }
  for (int off = 16; off > 0; off >>= 1) {
    s  += __shfl_xor(s, off, 32);
    ss += __shfl_xor(ss, off, 32);
  }
  __shared__ float sm0[4], sm1[4];
  if ((tid & 31) == 0) { sm0[tid >> 5] = s; sm1[tid >> 5] = ss; }
  __syncthreads();
  s  = sm0[0] + sm0[1] + sm0[2] + sm0[3];
  ss = sm1[0] + sm1[1] + sm1[2] + sm1[3];
  const float mu  = s * (1.0f / D_);
  const float var = ss * (1.0f / D_) - mu * mu;
  const float inv = rsqrtf(var + 1e-6f);
  _Float16* o = out + (size_t)row * D_;
#pragma unroll
  for (int i = tid; i < D_; i += 128)
    o[i] = (_Float16)((xr[i] - mu) * inv * g[i] + b[i]);
}

// ---------------- f32 -> f16 convert ----------------
__global__ void cvt_kernel(const float* __restrict__ in, _Float16* __restrict__ out, int n) {
  int i = blockIdx.x * blockDim.x + threadIdx.x;
  if (i < n) out[i] = (_Float16)in[i];
}

// ---------------- Projection GEMM: [2048,512] x [512,512]^T -> head-permuted f16 ----
// vmode 0: out[((b*H+h)*L + l)*64 + d]        (Q, K layout)
// vmode 1: out[((b*H+h)*64 + d)*L + l]        (V transposed layout for NT PV)
__global__ void __launch_bounds__(256)
proj_kernel(const _Float16* __restrict__ X, const _Float16* __restrict__ W,
            _Float16* __restrict__ out, float scale, int vmode) {
  __shared__ __align__(16) _Float16 lds[2 * 64 * 32];
  const int wave = threadIdx.x >> 5;     // 0..7
  const int lane = threadIdx.x & 31;
  const int m0 = (blockIdx.x * 8 + wave) * 16;   // 16 blocks x 8 waves = 128 m-tiles
  const int n0 = blockIdx.y * 64;                // 8 n-supertiles
  v8f acc[4] = {};
  gemm_nt_lds(X + (size_t)m0 * D_, D_, W + (size_t)n0 * D_, D_, D_, lds, acc);
  const int nl = lane & 15, mb = (lane >> 4) * 8;
#pragma unroll
  for (int j = 0; j < 4; ++j) {
    const int n = n0 + 16 * j + nl;
    const int h = n >> 6, d = n & 63;
#pragma unroll
    for (int v = 0; v < 8; ++v) {
      const int m = m0 + mb + v;
      const int bb = m >> 10, l = m & (L_ - 1);
      const size_t idx = vmode
          ? ((size_t)(bb * H_ + h) * DV_ + d) * L_ + l
          : ((size_t)(bb * H_ + h) * L_ + l) * DK_ + d;
      out[idx] = (_Float16)(acc[j][v] * scale);
    }
  }
}

// ---------------- Scores: per (b,h): S[L,L] = Qh[L,64] x Kh[L,64]^T, fp32 out ------
__global__ void __launch_bounds__(256)
scores_kernel(const _Float16* __restrict__ qh, const _Float16* __restrict__ kh,
              float* __restrict__ attn) {
  const int bh = blockIdx.z;
  const _Float16* Q  = qh + (size_t)bh * L_ * DK_;
  const _Float16* Kp = kh + (size_t)bh * L_ * DK_;
  float* S = attn + (size_t)bh * L_ * L_;
  const int wave = (blockIdx.x * 256 + threadIdx.x) >> 5;
  const int lane = threadIdx.x & 31;
  const int mt = wave % 64, nt = wave / 64;      // nt: 0..15
  const int m0 = mt * 16, n0 = nt * 64;
  v8f acc[4] = {};
#pragma unroll
  for (int k = 0; k < DK_; k += 32) {
    v16h a = frag_a(Q + (size_t)m0 * DK_ + k, DK_, lane);
#pragma unroll
    for (int j = 0; j < 4; ++j) {
      v16h b = frag_b_nt(Kp + (size_t)(n0 + 16 * j) * DK_ + k, DK_, lane);
      acc[j] = wmma16(a, b, acc[j]);
    }
  }
  const int nl = lane & 15, mb = (lane >> 4) * 8;
#pragma unroll
  for (int j = 0; j < 4; ++j)
#pragma unroll
    for (int v = 0; v < 8; ++v)
      S[(size_t)(m0 + mb + v) * L_ + n0 + 16 * j + nl] = acc[j][v];
}

// ---------------- Masked softmax row kernel -> f16 probs ----------------
__global__ void __launch_bounds__(256)
softmax_kernel(const float* __restrict__ attn, const float* __restrict__ mask,
               _Float16* __restrict__ probs) {
  const int row = blockIdx.x;                    // B*H*L rows
  const int tid = threadIdx.x;                   // 256, 4 elems each
  const int qrow = row & (L_ - 1);
  const float* s = attn + (size_t)row * L_;
  const float* m = mask + (size_t)qrow * L_;
  float vals[4];
  float mx = -3.4e38f;
#pragma unroll
  for (int i = 0; i < 4; ++i) {
    const int c = tid + i * 256;
    float v = s[c] + (1.0f - m[c]) * -10000.0f;
    vals[i] = v;
    mx = fmaxf(mx, v);
  }
  __shared__ float smx[8], ssum[8];
  for (int off = 16; off > 0; off >>= 1) mx = fmaxf(mx, __shfl_xor(mx, off, 32));
  if ((tid & 31) == 0) smx[tid >> 5] = mx;
  __syncthreads();
  mx = smx[0];
#pragma unroll
  for (int i = 1; i < 8; ++i) mx = fmaxf(mx, smx[i]);
  float sum = 0.f;
#pragma unroll
  for (int i = 0; i < 4; ++i) { vals[i] = __expf(vals[i] - mx); sum += vals[i]; }
  for (int off = 16; off > 0; off >>= 1) sum += __shfl_xor(sum, off, 32);
  if ((tid & 31) == 0) ssum[tid >> 5] = sum;
  __syncthreads();
  sum = ssum[0];
#pragma unroll
  for (int i = 1; i < 8; ++i) sum += ssum[i];
  const float inv = 1.0f / sum;
  _Float16* o = probs + (size_t)row * L_;
#pragma unroll
  for (int i = 0; i < 4; ++i) o[tid + i * 256] = (_Float16)(vals[i] * inv);
}

// ---------------- PV (NT): per (b,h): O[L,64] = probs[L,L] x Vt[64,L]^T ------------
// Vt layout: [B,H,DV,L]. Output permuted f16 -> ao[B, L, H*DV].
__global__ void __launch_bounds__(256)
pv_kernel(const _Float16* __restrict__ probs, const _Float16* __restrict__ vt,
          _Float16* __restrict__ ao) {
  __shared__ __align__(16) _Float16 lds[2 * 64 * 32];
  const int bh = blockIdx.z;
  const _Float16* Pm = probs + (size_t)bh * L_ * L_;
  const _Float16* Vt = vt + (size_t)bh * DV_ * L_;      // 64 rows x 1024
  const int bb = bh >> 3, h = bh & 7;
  const int wave = threadIdx.x >> 5;
  const int lane = threadIdx.x & 31;
  const int m0 = (blockIdx.x * 8 + wave) * 16;          // 8 blocks x 8 waves = 64 m-tiles
  v8f acc[4] = {};
  gemm_nt_lds(Pm + (size_t)m0 * L_, L_, Vt, L_, L_, lds, acc);
  const int nl = lane & 15, mb = (lane >> 4) * 8;
#pragma unroll
  for (int j = 0; j < 4; ++j) {
    const int d = 16 * j + nl;
#pragma unroll
    for (int v = 0; v < 8; ++v) {
      const int q = m0 + mb + v;
      ao[((size_t)(bb * L_ + q)) * (H_ * DV_) + h * DV_ + d] = (_Float16)acc[j][v];
    }
  }
}

// ---------------- FC + bias + residual: out = AO x fc_w^T + fc_b + q ----------------
__global__ void __launch_bounds__(256)
fc_kernel(const _Float16* __restrict__ A, const _Float16* __restrict__ W,
          const float* __restrict__ bias, const float* __restrict__ resid,
          float* __restrict__ out) {
  __shared__ __align__(16) _Float16 lds[2 * 64 * 32];
  const int wave = threadIdx.x >> 5;
  const int lane = threadIdx.x & 31;
  const int m0 = (blockIdx.x * 8 + wave) * 16;
  const int n0 = blockIdx.y * 64;
  v8f acc[4] = {};
  gemm_nt_lds(A + (size_t)m0 * D_, D_, W + (size_t)n0 * D_, D_, D_, lds, acc);
  const int nl = lane & 15, mb = (lane >> 4) * 8;
#pragma unroll
  for (int j = 0; j < 4; ++j) {
    const int n = n0 + 16 * j + nl;
#pragma unroll
    for (int v = 0; v < 8; ++v) {
      const int m = m0 + mb + v;
      out[(size_t)m * D_ + n] = acc[j][v] + bias[n] + resid[(size_t)m * D_ + n];
    }
  }
}

// ---------------- Host orchestration ----------------
extern "C" void kernel_launch(void* const* d_in, const int* in_sizes, int n_in,
                              void* d_out, int out_size, void* d_ws, size_t ws_size,
                              hipStream_t stream) {
  const float* q    = (const float*)d_in[0];
  const float* k    = (const float*)d_in[1];
  const float* v    = (const float*)d_in[2];
  const float* mask = (const float*)d_in[3];
  const float* w_q  = (const float*)d_in[4];
  const float* w_k  = (const float*)d_in[5];
  const float* w_v  = (const float*)d_in[6];
  const float* fc_w = (const float*)d_in[7];
  const float* fc_b = (const float*)d_in[8];
  const float* ln_g = (const float*)d_in[9];
  const float* ln_b = (const float*)d_in[10];

  float* out  = (float*)d_out;                             // [P,B,L,D]
  float* attn = out + (size_t)P_ * B_ * L_ * D_;           // [B,H,L,L]

  char* ws = (char*)d_ws;
  auto carve = [&](size_t bytes) { char* p = ws; ws += (bytes + 255) & ~(size_t)255; return p; };
  _Float16* qn  = (_Float16*)carve((size_t)B_ * L_ * D_ * 2);
  _Float16* kb  = (_Float16*)carve((size_t)B_ * L_ * D_ * 2);
  _Float16* vb  = (_Float16*)carve((size_t)B_ * L_ * D_ * 2);
  _Float16* wqb = (_Float16*)carve((size_t)D_ * D_ * 2);
  _Float16* wkb = (_Float16*)carve((size_t)D_ * D_ * 2);
  _Float16* wvb = (_Float16*)carve((size_t)D_ * D_ * 2);
  _Float16* fwb = (_Float16*)carve((size_t)D_ * D_ * 2);
  _Float16* qh  = (_Float16*)carve((size_t)B_ * H_ * L_ * DK_ * 2);
  _Float16* kh  = (_Float16*)carve((size_t)B_ * H_ * L_ * DK_ * 2);
  _Float16* vt  = (_Float16*)carve((size_t)B_ * H_ * DV_ * L_ * 2);
  _Float16* probs = (_Float16*)carve((size_t)B_ * H_ * L_ * L_ * 2);
  _Float16* ao  = (_Float16*)carve((size_t)B_ * L_ * H_ * DV_ * 2);

  // Stage 1: LN(q) + f16 conversions
  ln_kernel<<<B_ * L_, 128, 0, stream>>>(q, ln_g, ln_b, qn);
  const int nAct = B_ * L_ * D_;
  const int nW = D_ * D_;
  cvt_kernel<<<(nAct + 255) / 256, 256, 0, stream>>>(k, kb, nAct);
  cvt_kernel<<<(nAct + 255) / 256, 256, 0, stream>>>(v, vb, nAct);
  cvt_kernel<<<(nW + 255) / 256, 256, 0, stream>>>(w_q, wqb, nW);
  cvt_kernel<<<(nW + 255) / 256, 256, 0, stream>>>(w_k, wkb, nW);
  cvt_kernel<<<(nW + 255) / 256, 256, 0, stream>>>(w_v, wvb, nW);
  cvt_kernel<<<(nW + 255) / 256, 256, 0, stream>>>(fc_w, fwb, nW);

  // Stage 2: projections (scale 1/sqrt(dk) folded into Q; V stored transposed)
  const float invT = 1.0f / sqrtf((float)DK_);
  proj_kernel<<<dim3(16, 8), 256, 0, stream>>>(qn, wqb, qh, invT, 0);
  proj_kernel<<<dim3(16, 8), 256, 0, stream>>>(kb, wkb, kh, 1.0f, 0);
  proj_kernel<<<dim3(16, 8), 256, 0, stream>>>(vb, wvb, vt, 1.0f, 1);

  // Stage 3: attention scores -> second output region (fp32)
  scores_kernel<<<dim3(128, 1, B_ * H_), 256, 0, stream>>>(qh, kh, attn);

  // Stage 4: per-mask softmax -> PV -> FC(+bias+residual)
  for (int p = 0; p < P_; ++p) {
    softmax_kernel<<<B_ * H_ * L_, 256, 0, stream>>>(attn, mask + (size_t)p * L_ * L_, probs);
    pv_kernel<<<dim3(8, 1, B_ * H_), 256, 0, stream>>>(probs, vt, ao);
    fc_kernel<<<dim3(16, 8), 256, 0, stream>>>(ao, fwb, fc_b, q, out + (size_t)p * B_ * L_ * D_);
  }
}
